// SimpleAttentionBlock_21431886807689
// MI455X (gfx1250) — matrix operand order repaired and checked by reading, not compiled, not run on hardware
//
#include <hip/hip_runtime.h>
#include <hip/hip_bf16.h>
#include <math.h>

// MI455X / gfx1250: wave32, WMMA 16x16x32 bf16 (f32 accum), async global->LDS copies.
typedef __attribute__((ext_vector_type(16))) __bf16 v16bf;
typedef __attribute__((ext_vector_type(8)))  float  v8f;

#define B_  4
#define C_  256
#define N_  4096   // 64*64
#define G_  32

#define WMMA_BF16(a, b, c) \
  __builtin_amdgcn_wmma_f32_16x16x32_bf16(false, (a), false, (b), (short)0, (c), false, false)

static __device__ __forceinline__ __bf16 f2bf(float f) {
  union { float f; unsigned u; } a; a.f = f;
  unsigned r = a.u + 0x7FFFu + ((a.u >> 16) & 1u);  // round-to-nearest-even
  unsigned short h = (unsigned short)(r >> 16);
  __bf16 o; __builtin_memcpy(&o, &h, 2); return o;
}

// Async global->LDS copy, 16B per lane. VDST = VGPR holding LDS byte offset
// (low 32 bits of a generic LDS pointer are the wave-relative LDS address),
// VADDR = 64-bit global address. Tracked by ASYNCcnt.
static __device__ __forceinline__ void async_ld_b128(__bf16* lds_dst, const __bf16* gsrc) {
  unsigned loff = (unsigned)(unsigned long long)lds_dst;
  asm volatile("global_load_async_to_lds_b128 %0, %1, off"
               :: "v"(loff), "v"(gsrc)
               : "memory");
}
static __device__ __forceinline__ void wait_async() {
  asm volatile("s_wait_asynccnt 0x0" ::: "memory");
}

// A-fragment (16x32 bf16, MxK): lane m=lane&15 holds K {0..7,16..23} (lanes<16)
// or {8..15,24..31} (lanes>=16); 2 contiguous 16B LDS loads.
static __device__ __forceinline__ v16bf lds_frag_a(const __bf16* tile, int ldh, int k0) {
  const int lane = threadIdx.x & 31;
  const __bf16* p = tile + (lane & 15) * ldh + k0 + ((lane & 16) ? 8 : 0);
  union { uint4 u[2]; v16bf v; } u;
  u.u[0] = *(const uint4*)(p);
  u.u[1] = *(const uint4*)(p + 16);
  return u.v;
}
// B-fragment (32x16 bf16, KxN): lane n=lane&15 holds K {0..7,16..23} (lanes<16) or
// {8..15,24..31} (lanes>=16) of column n == row n of W (row-major [N,K] tile).
static __device__ __forceinline__ v16bf lds_frag_b(const __bf16* tile, int ldh, int k0) {
  const int lane = threadIdx.x & 31;
  const __bf16* p = tile + (lane & 15) * ldh + k0 + ((lane & 16) ? 8 : 0);
  union { uint4 u[2]; v16bf v; } u;
  u.u[0] = *(const uint4*)(p);
  u.u[1] = *(const uint4*)(p + 16);
  return u.v;
}

// ---------------- fp32 -> bf16 weight conversion ----------------
__global__ __launch_bounds__(256) void cvt_bf16_kernel(const float* __restrict__ in,
                                                       __bf16* __restrict__ out, int n) {
  int i = blockIdx.x * 256 + threadIdx.x;
  if (i < n) out[i] = f2bf(in[i]);
}

// ---------------- GroupNorm: [B,C,N] fp32 -> token-major [B,N,C] bf16 --------
__global__ __launch_bounds__(256) void groupnorm_kernel(const float* __restrict__ x,
                                                        const float* __restrict__ gamma,
                                                        const float* __restrict__ beta,
                                                        __bf16* __restrict__ xn) {
  const int g = blockIdx.x, b = blockIdx.y, t = threadIdx.x;
  const size_t base = ((size_t)b * C_ + g * 8) * N_;  // 8 chans * 4096 contiguous
  const float4* xv = (const float4*)(x + base);
  float s1 = 0.f, s2 = 0.f;
  #pragma unroll 4
  for (int i = t; i < 8192; i += 256) {
    float4 v = xv[i];
    s1 += v.x + v.y + v.z + v.w;
    s2 += v.x * v.x + v.y * v.y + v.z * v.z + v.w * v.w;
  }
  __shared__ float r1[256], r2[256];
  r1[t] = s1; r2[t] = s2;
  __syncthreads();
  for (int off = 128; off > 0; off >>= 1) {
    if (t < off) { r1[t] += r1[t + off]; r2[t] += r2[t + off]; }
    __syncthreads();
  }
  const float inv  = 1.0f / 32768.0f;
  const float mean = r1[0] * inv;
  const float var  = r2[0] * inv - mean * mean;
  const float rstd = rsqrtf(var + 1e-5f);
  const int   cl   = t & 7;
  const float gm   = gamma[g * 8 + cl] * rstd;
  const float sh   = beta[g * 8 + cl] - mean * gm;
  __bf16* xno = xn + (size_t)b * N_ * C_ + g * 8 + cl;
  const float* xr = x + base + (size_t)cl * N_;
  for (int it = 0; it < 128; ++it) {
    int n = it * 32 + (t >> 3);
    xno[(size_t)n * C_] = f2bf(xr[n] * gm + sh);
  }
}

// ---------------- QKV GEMM: [N,256] x [768,256]^T, 128x128 block tiles -------
// Double-buffered async LDS staging: copy slice k+1 while WMMAs run on slice k.
__global__ __launch_bounds__(256) void qkv_gemm_kernel(const __bf16* __restrict__ xn,
                                                       const __bf16* __restrict__ wq,
                                                       __bf16* __restrict__ q,
                                                       __bf16* __restrict__ k,
                                                       __bf16* __restrict__ vt) {
  __shared__ __bf16 As[2][128 * 40];
  __shared__ __bf16 Bs[2][128 * 40];
  const int t = threadIdx.x, lane = t & 31, wid = t >> 5;
  const int wm = wid >> 1, wn = wid & 1;
  const int b = blockIdx.z;
  const int t0 = blockIdx.x * 128;  // token base
  const int o0 = blockIdx.y * 128;  // out-channel base
  const __bf16* Ag = xn + (size_t)b * N_ * C_ + (size_t)t0 * C_;
  const __bf16* Bg = wq + (size_t)o0 * C_;

  auto stage = [&](int buf, int k0) {
    #pragma unroll
    for (int s = 0; s < 2; ++s) {
      int ch = t * 2 + s;                       // 512 chunks of 8 halves
      int row = ch >> 2, c8 = (ch & 3) * 8;
      async_ld_b128(&As[buf][row * 40 + c8], &Ag[(size_t)row * C_ + k0 + c8]);
      async_ld_b128(&Bs[buf][row * 40 + c8], &Bg[(size_t)row * C_ + k0 + c8]);
    }
  };

  v8f acc[2][4];
  #pragma unroll
  for (int i = 0; i < 2; ++i)
    #pragma unroll
    for (int j = 0; j < 4; ++j)
      #pragma unroll
      for (int e = 0; e < 8; ++e) acc[i][j][e] = 0.f;

  stage(0, 0);
  wait_async();
  __syncthreads();

  for (int kt = 0; kt < 8; ++kt) {
    const int cur = kt & 1;
    if (kt + 1 < 8) stage(cur ^ 1, (kt + 1) * 32);
    v16bf a0 = lds_frag_a(&As[cur][(wm * 32) * 40], 40, 0);
    v16bf a1 = lds_frag_a(&As[cur][(wm * 32 + 16) * 40], 40, 0);
    #pragma unroll
    for (int j = 0; j < 4; ++j) {
      v16bf bb = lds_frag_b(&Bs[cur][(wn * 64 + j * 16) * 40], 40, 0);
      acc[0][j] = WMMA_BF16(a0, bb, acc[0][j]);
      acc[1][j] = WMMA_BF16(a1, bb, acc[1][j]);
    }
    wait_async();
    __syncthreads();
  }

  const int hl = lane >> 4, col = lane & 15;
  const int sel = blockIdx.y >> 1;  // 0=Q, 1=K, 2=V
  #pragma unroll
  for (int i = 0; i < 2; ++i)
    #pragma unroll
    for (int j = 0; j < 4; ++j)
      #pragma unroll
      for (int r = 0; r < 8; ++r) {
        int tok = t0 + wm * 32 + i * 16 + hl * 8 + r;
        int oc  = o0 + wn * 64 + j * 16 + col;
        float v = acc[i][j][r];
        if (sel == 0)       q[((size_t)b * N_ + tok) * C_ + oc]        = f2bf(v * 0.0625f); // fold 1/sqrt(C)
        else if (sel == 1)  k[((size_t)b * N_ + tok) * C_ + (oc - 256)] = f2bf(v);
        else                vt[((size_t)b * C_ + (oc - 512)) * N_ + tok] = f2bf(v); // V channel-major
      }
}

// ---------------- Flash attention: 128-query tiles, 64-key tiles -------------
// Q tile -> register A-fragments; K/V tiles double-buffered via async copies.
// smem: Ks[2] 64x264, Vs[2] 256x72, Ps 128x72; Q staging aliases buffer region.
#define ATTN_SMEM_HALVES (2 * 64 * 264 + 2 * 256 * 72 + 128 * 72)
__global__ __launch_bounds__(256) void attn_kernel(const __bf16* __restrict__ q,
                                                   const __bf16* __restrict__ kmat,
                                                   const __bf16* __restrict__ vt,
                                                   __bf16* __restrict__ o) {
  extern __shared__ __bf16 smem[];
  __bf16* Ks0 = smem;                    // 16896 halves
  __bf16* Ks1 = Ks0 + 64 * 264;          // 16896
  __bf16* Vs0 = Ks1 + 64 * 264;          // 18432
  __bf16* Vs1 = Vs0 + 256 * 72;          // 18432
  __bf16* Ps  = Vs1 + 256 * 72;          // 9216
  __bf16* Qs  = smem;                    // staging alias: 128*264 = 33792 halves
  const int t = threadIdx.x, lane = t & 31, w = t >> 5;
  const int b = blockIdx.y, q0 = blockIdx.x * 128;

  // Stage Q tile once (async), capture per-wave A-fragments into registers.
  const __bf16* Qg = q + ((size_t)b * N_ + q0) * C_;
  #pragma unroll
  for (int it = 0; it < 16; ++it) {
    int ch = t + it * 256;                    // 4096 chunks of 8 halves
    int row = ch >> 5, c8 = (ch & 31) * 8;
    async_ld_b128(&Qs[row * 264 + c8], &Qg[(size_t)row * C_ + c8]);
  }
  wait_async();
  __syncthreads();
  v16bf aq[8];
  #pragma unroll
  for (int i = 0; i < 8; ++i) aq[i] = lds_frag_a(&Qs[(w * 16) * 264], 264, i * 32);
  __syncthreads();  // Q frags live in VGPRs; smem free for K/V buffers

  float Mrow[8], Lrow[8];
  #pragma unroll
  for (int r = 0; r < 8; ++r) { Mrow[r] = -1e30f; Lrow[r] = 0.f; }
  v8f oacc[16];
  #pragma unroll
  for (int f = 0; f < 16; ++f)
    #pragma unroll
    for (int e = 0; e < 8; ++e) oacc[f][e] = 0.f;

  const __bf16* Kg = kmat + (size_t)b * N_ * C_;
  const __bf16* Vg = vt + (size_t)b * C_ * N_;
  const int hl = lane >> 4, col = lane & 15;

  auto stage_kv = [&](int buf, int j0) {
    __bf16* Kd = buf ? Ks1 : Ks0;
    __bf16* Vd = buf ? Vs1 : Vs0;
    #pragma unroll
    for (int it = 0; it < 8; ++it) {          // K tile: 64x256 -> [64][264]
      int ch = t + it * 256;
      int row = ch >> 5, c8 = (ch & 31) * 8;
      async_ld_b128(&Kd[row * 264 + c8], &Kg[(size_t)(j0 + row) * C_ + c8]);
    }
    #pragma unroll
    for (int it = 0; it < 8; ++it) {          // V^T tile: 256x64 -> [256][72]
      int ch = t + it * 256;
      int row = ch >> 3, c8 = (ch & 7) * 8;
      async_ld_b128(&Vd[row * 72 + c8], &Vg[(size_t)row * N_ + j0 + c8]);
    }
  };

  stage_kv(0, 0);
  wait_async();
  __syncthreads();

  for (int jt = 0; jt < N_ / 64; ++jt) {
    const int cur = jt & 1;
    if (jt + 1 < N_ / 64) stage_kv(cur ^ 1, (jt + 1) * 64);  // prefetch next tile
    const __bf16* Kc = cur ? Ks1 : Ks0;
    const __bf16* Vc = cur ? Vs1 : Vs0;

    // S = Q K^T  (per wave: 16 rows x 64 keys)
    v8f s[4];
    #pragma unroll
    for (int j = 0; j < 4; ++j)
      #pragma unroll
      for (int e = 0; e < 8; ++e) s[j][e] = 0.f;
    #pragma unroll
    for (int k0 = 0; k0 < 8; ++k0)
      #pragma unroll
      for (int j = 0; j < 4; ++j) {
        v16bf bk = lds_frag_b(&Kc[(j * 16) * 264], 264, k0 * 32);
        s[j] = WMMA_BF16(aq[k0], bk, s[j]);
      }

    // Online softmax; row = (hl*8 + r), columns across 16-lane halves.
    #pragma unroll
    for (int r = 0; r < 8; ++r) {
      float mx = fmaxf(fmaxf(s[0][r], s[1][r]), fmaxf(s[2][r], s[3][r]));
      #pragma unroll
      for (int off = 8; off > 0; off >>= 1) mx = fmaxf(mx, __shfl_xor(mx, off, 32));
      float nm = fmaxf(Mrow[r], mx);
      float alpha = __expf(Mrow[r] - nm);
      Mrow[r] = nm;
      float rs = 0.f;
      #pragma unroll
      for (int j = 0; j < 4; ++j) {
        float p = __expf(s[j][r] - nm);
        s[j][r] = p; rs += p;
      }
      #pragma unroll
      for (int off = 8; off > 0; off >>= 1) rs += __shfl_xor(rs, off, 32);
      Lrow[r] = Lrow[r] * alpha + rs;
      #pragma unroll
      for (int f = 0; f < 16; ++f) oacc[f][r] *= alpha;
      int prow = w * 16 + hl * 8 + r;
      #pragma unroll
      for (int j = 0; j < 4; ++j) Ps[prow * 72 + j * 16 + col] = f2bf(s[j][r]);
    }
    __syncthreads();

    // O += P V  (A = P tile 16x64, B = V^T tiles)
    v16bf ap0 = lds_frag_a(&Ps[(w * 16) * 72], 72, 0);
    v16bf ap1 = lds_frag_a(&Ps[(w * 16) * 72], 72, 32);
    #pragma unroll
    for (int f = 0; f < 16; ++f) {
      v16bf bv0 = lds_frag_b(&Vc[(f * 16) * 72], 72, 0);
      v16bf bv1 = lds_frag_b(&Vc[(f * 16) * 72], 72, 32);
      oacc[f] = WMMA_BF16(ap0, bv0, oacc[f]);
      oacc[f] = WMMA_BF16(ap1, bv1, oacc[f]);
    }
    wait_async();      // next tile's async copies landed
    __syncthreads();
  }

  // Normalize and store O token-major bf16 [B,N,C]
  __bf16* Og = o + ((size_t)b * N_ + q0) * C_;
  #pragma unroll
  for (int f = 0; f < 16; ++f)
    #pragma unroll
    for (int r = 0; r < 8; ++r) {
      int tok = w * 16 + hl * 8 + r;
      Og[(size_t)tok * C_ + f * 16 + col] = f2bf(oacc[f][r] / Lrow[r]);
    }
}

// ---------------- Proj GEMM + residual: out[b,c,n] = x + O @ Wp^T ------------
__global__ __launch_bounds__(256) void proj_gemm_kernel(const __bf16* __restrict__ o,
                                                        const __bf16* __restrict__ wp,
                                                        const float* __restrict__ x,
                                                        float* __restrict__ out) {
  __shared__ __bf16 As[2][128 * 40];
  __shared__ __bf16 Bs[2][128 * 40];
  const int t = threadIdx.x, lane = t & 31, wid = t >> 5;
  const int wm = wid >> 1, wn = wid & 1;
  const int b = blockIdx.z;
  const int t0 = blockIdx.x * 128;
  const int o0 = blockIdx.y * 128;
  const __bf16* Ag = o + (size_t)b * N_ * C_ + (size_t)t0 * C_;
  const __bf16* Bg = wp + (size_t)o0 * C_;

  auto stage = [&](int buf, int k0) {
    #pragma unroll
    for (int s = 0; s < 2; ++s) {
      int ch = t * 2 + s;
      int row = ch >> 2, c8 = (ch & 3) * 8;
      async_ld_b128(&As[buf][row * 40 + c8], &Ag[(size_t)row * C_ + k0 + c8]);
      async_ld_b128(&Bs[buf][row * 40 + c8], &Bg[(size_t)row * C_ + k0 + c8]);
    }
  };

  v8f acc[2][4];
  #pragma unroll
  for (int i = 0; i < 2; ++i)
    #pragma unroll
    for (int j = 0; j < 4; ++j)
      #pragma unroll
      for (int e = 0; e < 8; ++e) acc[i][j][e] = 0.f;

  stage(0, 0);
  wait_async();
  __syncthreads();

  for (int kt = 0; kt < 8; ++kt) {
    const int cur = kt & 1;
    if (kt + 1 < 8) stage(cur ^ 1, (kt + 1) * 32);
    v16bf a0 = lds_frag_a(&As[cur][(wm * 32) * 40], 40, 0);
    v16bf a1 = lds_frag_a(&As[cur][(wm * 32 + 16) * 40], 40, 0);
    #pragma unroll
    for (int j = 0; j < 4; ++j) {
      v16bf bb = lds_frag_b(&Bs[cur][(wn * 64 + j * 16) * 40], 40, 0);
      acc[0][j] = WMMA_BF16(a0, bb, acc[0][j]);
      acc[1][j] = WMMA_BF16(a1, bb, acc[1][j]);
    }
    wait_async();
    __syncthreads();
  }

  const int hl = lane >> 4, col = lane & 15;
  #pragma unroll
  for (int i = 0; i < 2; ++i)
    #pragma unroll
    for (int j = 0; j < 4; ++j)
      #pragma unroll
      for (int r = 0; r < 8; ++r) {
        int tok = t0 + wm * 32 + i * 16 + hl * 8 + r;
        int oc  = o0 + wn * 64 + j * 16 + col;
        size_t idx = ((size_t)b * C_ + oc) * N_ + tok;  // output is [B,C,H,W]
        out[idx] = x[idx] + acc[i][j][r];
      }
}

// -----------------------------------------------------------------------------
extern "C" void kernel_launch(void* const* d_in, const int* in_sizes, int n_in,
                              void* d_out, int out_size, void* d_ws, size_t ws_size,
                              hipStream_t stream) {
  const float* x     = (const float*)d_in[0];
  // d_in[1] = t (unused scalar)
  const float* gamma = (const float*)d_in[2];
  const float* beta  = (const float*)d_in[3];
  const float* wqkv  = (const float*)d_in[4];
  const float* wproj = (const float*)d_in[5];
  float* out = (float*)d_out;

  char* ws = (char*)d_ws;
  size_t off = 0;
  auto carve = [&](size_t bytes) -> void* {
    void* p = ws + off;
    off = (off + bytes + 255) & ~(size_t)255;
    return p;
  };
  const size_t act_bytes = (size_t)B_ * N_ * C_ * sizeof(__bf16);  // 8 MiB each
  __bf16* xn  = (__bf16*)carve(act_bytes);
  __bf16* qb  = (__bf16*)carve(act_bytes);
  __bf16* kb  = (__bf16*)carve(act_bytes);
  __bf16* vtb = (__bf16*)carve(act_bytes);   // V stored [B,C,N]
  __bf16* ob  = (__bf16*)carve(act_bytes);
  __bf16* wqb = (__bf16*)carve((size_t)3 * C_ * C_ * sizeof(__bf16));
  __bf16* wpb = (__bf16*)carve((size_t)C_ * C_ * sizeof(__bf16));

  cvt_bf16_kernel<<<(3 * C_ * C_ + 255) / 256, 256, 0, stream>>>(wqkv, wqb, 3 * C_ * C_);
  cvt_bf16_kernel<<<(C_ * C_ + 255) / 256, 256, 0, stream>>>(wproj, wpb, C_ * C_);
  groupnorm_kernel<<<dim3(G_, B_), 256, 0, stream>>>(x, gamma, beta, xn);
  qkv_gemm_kernel<<<dim3(N_ / 128, 6, B_), 256, 0, stream>>>(xn, wqb, qb, kb, vtb);
  attn_kernel<<<dim3(N_ / 128, B_), 256, ATTN_SMEM_HALVES * sizeof(__bf16), stream>>>(qb, kb, vtb, ob);
  proj_gemm_kernel<<<dim3(N_ / 128, C_ / 128, B_), 256, 0, stream>>>(ob, wpb, x, out);
}
